// MinkEncoderDecoder_45990509805970
// MI455X (gfx1250) — matrix operand order repaired and checked by reading, not compiled, not run on hardware
//
#include <hip/hip_runtime.h>
#include <hip/hip_bf16.h>

typedef _Float16 half_t;
typedef __attribute__((ext_vector_type(16))) _Float16 v16h;
typedef __attribute__((ext_vector_type(8)))  _Float16 v8h;
typedef __attribute__((ext_vector_type(8)))  float    v8f;

// ---------------------------------------------------------------------------
// WMMA gather-GEMM:  out[rows, cout] = gather(A16) @ Wt
//   A16 : [rows_in, cinP] f16 activations (cinP = ceil32(cin), zero padded)
//   Wt  : [K, coutP, cinP] f16 weights (transposed so K-dim is contiguous)
//   nbr : conv mode    -> [rows, K] neighbor indices (-1 = missing)
//   koff: deconv mode  -> src = (koff[row]==k) ? nbr[row] : -1   (nbr=parent)
//   nbr==null          -> identity (dense GEMM, K==1)
//
// Block = 4 waves (32,4). Each wave owns a 16-row tile and a 16x32 output
// tile (two v8f accumulators -> two v_wmma_f32_16x16x32_f16 per K-chunk).
// The 32-column weight panel for the current k is staged into LDS once per
// block (shared by all 4 waves), read back via ds_load_b128.
// ---------------------------------------------------------------------------
#define BP_COLS 32
#define BP_CINP 384   // max padded cin (decoder concat 256+128)

__global__ void wmma_gemm_kernel(const half_t* __restrict__ A16,
                                 const int*    __restrict__ nbr,
                                 const int*    __restrict__ koff,
                                 const half_t* __restrict__ Wt,
                                 float*        __restrict__ out,
                                 int rows, int cinP, int cout, int coutP, int K)
{
    __shared__ __align__(16) half_t bpanel[BP_COLS * BP_CINP];

    const int lane  = threadIdx.x;          // 0..31
    const int tid   = threadIdx.y * 32 + threadIdx.x;
    const int tile  = blockIdx.x * blockDim.y + threadIdx.y;
    const int mrow  = tile * 16 + (lane & 15);
    const int n0    = blockIdx.y * 32;
    const int col0  = n0 + (lane & 15);
    const int col1  = col0 + 16;
    const bool hi   = lane >= 16;
    const int  c8   = cinP >> 3;            // v8h chunks per panel column

    v8f acc0 = {};
    v8f acc1 = {};

    for (int k = 0; k < K; ++k) {
        int src = -1;
        if (mrow < rows) {
            if (koff)      src = (koff[mrow] == k) ? nbr[mrow] : -1;
            else if (nbr)  src = nbr[(long long)mrow * K + k];
            else           src = mrow;
        }
        const half_t* arow = (src >= 0) ? (A16 + (long long)src * cinP) : nullptr;

        // ---- stage 32-column weight panel for this k into LDS (block-wide) ----
        __syncthreads();   // previous k's LDS reads done
        for (int t = tid; t < BP_COLS * c8; t += 128) {
            int cl = t / c8;
            int o8 = t - cl * c8;
            *(v8h*)(bpanel + cl * cinP + o8 * 8) =
                *(const v8h*)(Wt + ((long long)k * coutP + n0 + cl) * cinP + o8 * 8);
        }
        if (k + 1 < K)     // gfx1250 global_prefetch_b8 on next weight panel
            __builtin_prefetch(Wt + ((long long)(k + 1) * coutP + n0 + (tid & 31)) * cinP, 0, 1);
        __syncthreads();

        const half_t* bl0 = bpanel + ((lane & 15) + 0)  * cinP + (hi ? 16 : 0);
        const half_t* bl1 = bpanel + ((lane & 15) + 16) * cinP + (hi ? 16 : 0);

        for (int c0 = 0; c0 < cinP; c0 += 32) {
            // A 16x32 f16, contiguous k order used here (same as B): lane<16 -> K c0+0..15
            //                     lane>=16 -> K c0+16..31
            v16h a;
            if (arow) {
                v8h alo = *(const v8h*)(arow + c0 + (hi ? 16 : 0));
                v8h ahi = *(const v8h*)(arow + c0 + (hi ? 24 : 8));
#pragma unroll
                for (int j = 0; j < 8; ++j) { a[j] = alo[j]; a[8 + j] = ahi[j]; }
            } else {
#pragma unroll
                for (int j = 0; j < 16; ++j) a[j] = (half_t)0.f;
            }
            // B 32x16 f16, contiguous k order used here (same as A): lane(&15)=N, lane<16 -> K c0..c0+15,
            //                     lane>=16 -> K c0+16..c0+31 (ptr pre-offset)
            v8h b0lo = *(const v8h*)(bl0 + c0);
            v8h b0hi = *(const v8h*)(bl0 + c0 + 8);
            v8h b1lo = *(const v8h*)(bl1 + c0);
            v8h b1hi = *(const v8h*)(bl1 + c0 + 8);
            v16h b0, b1;
#pragma unroll
            for (int j = 0; j < 8; ++j) {
                b0[j] = b0lo[j]; b0[8 + j] = b0hi[j];
                b1[j] = b1lo[j]; b1[8 + j] = b1hi[j];
            }
            acc0 = __builtin_amdgcn_wmma_f32_16x16x32_f16(
                       false, a, false, b0, (short)0, acc0, false, false);
            acc1 = __builtin_amdgcn_wmma_f32_16x16x32_f16(
                       false, a, false, b1, (short)0, acc1, false, false);
        }
    }

    // C/D layout: lane<16 -> rows M=0..7, lane>=16 -> rows M=8..15, col = lane&15
    const int mbase = tile * 16 + (hi ? 8 : 0);
    if (col0 < cout) {
#pragma unroll
        for (int r = 0; r < 8; ++r) {
            int m = mbase + r;
            if (m < rows) out[(long long)m * cout + col0] = acc0[r];
        }
    }
    if (col1 < cout) {
#pragma unroll
        for (int r = 0; r < 8; ++r) {
            int m = mbase + r;
            if (m < rows) out[(long long)m * cout + col1] = acc1[r];
        }
    }
}

// ---------------------------------------------------------------------------
// Support kernels
// ---------------------------------------------------------------------------
__global__ void zero_kernel(float* p, long long n)
{
    long long i = (long long)blockIdx.x * blockDim.x + threadIdx.x;
    if (i < n) p[i] = 0.f;
}

__global__ void scatter_mean_kernel(const float* __restrict__ pf,
                                    const int*   __restrict__ p2v,
                                    float* __restrict__ v, int npts, int dim)
{
    int i = blockIdx.x * blockDim.x + threadIdx.x;
    if (i >= npts) return;
    int t = p2v[i];
    for (int c = 0; c < dim; ++c)
        atomicAdd(&v[(long long)t * dim + c], pf[(long long)i * dim + c]);
}

__global__ void divide_kernel(float* v, const float* __restrict__ counts,
                              long long n, int dim)
{
    long long i = (long long)blockIdx.x * blockDim.x + threadIdx.x;
    if (i < n) v[i] /= counts[i / dim];
}

__global__ void conv_a16_kernel(const float* __restrict__ in, half_t* __restrict__ out,
                                int rows, int cin, int cinP)
{
    long long i = (long long)blockIdx.x * blockDim.x + threadIdx.x;
    if (i >= (long long)rows * cinP) return;
    int col = (int)(i % cinP);
    long long r = i / cinP;
    out[i] = (col < cin) ? (half_t)in[r * cin + col] : (half_t)0.f;
}

// W f32 [K, cin, cout]  ->  Wt f16 [K, coutP, cinP]  (transposed, zero padded)
__global__ void conv_w16_kernel(const float* __restrict__ W, half_t* __restrict__ Wt,
                                int K, int cin, int cout, int cinP, int coutP)
{
    long long i = (long long)blockIdx.x * blockDim.x + threadIdx.x;
    long long tot = (long long)K * coutP * cinP;
    if (i >= tot) return;
    int c = (int)(i % cinP);
    long long t = i / cinP;
    int d = (int)(t % coutP);
    int k = (int)(t / coutP);
    Wt[i] = (c < cin && d < cout) ? (half_t)W[((long long)k * cin + c) * cout + d]
                                  : (half_t)0.f;
}

// Per-column mean/biased-var over rows (LDS tree reduction)
__global__ void col_stats_kernel(const float* __restrict__ in, int rows, int c,
                                 float* __restrict__ stats)
{
    __shared__ float ssum[256];
    __shared__ float ssq[256];
    int col = blockIdx.x;
    float s = 0.f, q = 0.f;
    for (int r = threadIdx.x; r < rows; r += blockDim.x) {
        float v = in[(long long)r * c + col];
        s += v; q += v * v;
    }
    ssum[threadIdx.x] = s; ssq[threadIdx.x] = q;
    __syncthreads();
    for (int off = 128; off > 0; off >>= 1) {
        if ((int)threadIdx.x < off) {
            ssum[threadIdx.x] += ssum[threadIdx.x + off];
            ssq[threadIdx.x]  += ssq[threadIdx.x + off];
        }
        __syncthreads();
    }
    if (threadIdx.x == 0) {
        float m = ssum[0] / (float)rows;
        stats[col]     = m;
        stats[c + col] = ssq[0] / (float)rows - m * m;
    }
}

// act: 0=none, 1=relu, 2=leaky(0.01); optional residual added pre-activation
__global__ void bn_apply_kernel(const float* __restrict__ in,
                                const float* __restrict__ stats,
                                const float* __restrict__ g,
                                const float* __restrict__ b,
                                const float* __restrict__ resid,
                                float* __restrict__ out,
                                int rows, int c, int act)
{
    long long i = (long long)blockIdx.x * blockDim.x + threadIdx.x;
    if (i >= (long long)rows * c) return;
    int col = (int)(i % c);
    float val = (in[i] - stats[col]) * rsqrtf(stats[c + col] + 1e-5f) * g[col] + b[col];
    if (resid) val += resid[i];
    if (act == 1)      val = fmaxf(val, 0.f);
    else if (act == 2) val = val > 0.f ? val : 0.01f * val;
    out[i] = val;
}

__global__ void copy_cols_kernel(const float* __restrict__ in, float* __restrict__ out,
                                 int rows, int cin, int ostride, int ooff)
{
    long long i = (long long)blockIdx.x * blockDim.x + threadIdx.x;
    if (i >= (long long)rows * cin) return;
    int col = (int)(i % cin);
    long long r = i / cin;
    out[r * ostride + ooff + col] = in[i];
}

// 3-NN inverse-distance interpolation (brute force, per batch)
__global__ void knn_interp_kernel(const float* __restrict__ ptc,
                                  const float* __restrict__ vm,
                                  const float* __restrict__ vf,
                                  const int*   __restrict__ segidx,
                                  const unsigned char* __restrict__ segmask,
                                  int Nloc, int Mmax, int C, int Btot,
                                  float* __restrict__ g)
{
    int gid = blockIdx.x * blockDim.x + threadIdx.x;
    if (gid >= Btot * Nloc) return;
    int b = gid / Nloc;
    const float* p = ptc + (long long)gid * 3;
    float px = p[0], py = p[1], pz = p[2];
    float d0 = 1e10f, d1 = 1e10f, d2 = 1e10f;
    int i0 = 0, i1 = 0, i2 = 0;
    const int* idxb = segidx + (long long)b * Mmax;
    const unsigned char* mb = segmask + (long long)b * Mmax;
    for (int m = 0; m < Mmax; ++m) {
        if (!mb[m]) continue;
        int vi = idxb[m];
        float dx = px - vm[(long long)vi * 3 + 0];
        float dy = py - vm[(long long)vi * 3 + 1];
        float dz = pz - vm[(long long)vi * 3 + 2];
        float d = dx * dx + dy * dy + dz * dz;
        if (d < d0)      { d2 = d1; i2 = i1; d1 = d0; i1 = i0; d0 = d; i0 = vi; }
        else if (d < d1) { d2 = d1; i2 = i1; d1 = d;  i1 = vi; }
        else if (d < d2) { d2 = d;  i2 = vi; }
    }
    float w0 = 1.f / (sqrtf(d0) + 1e-8f);
    float w1 = 1.f / (sqrtf(d1) + 1e-8f);
    float w2 = 1.f / (sqrtf(d2) + 1e-8f);
    float ws = w0 + w1 + w2;
    w0 /= ws; w1 /= ws; w2 /= ws;
    const float* f0 = vf + (long long)i0 * C;
    const float* f1 = vf + (long long)i1 * C;
    const float* f2 = vf + (long long)i2 * C;
    float* go = g + (long long)gid * C;
    for (int c = 0; c < C; ++c)
        go[c] = w0 * f0[c] + w1 * f1[c] + w2 * f2[c];
}

__global__ void coors_copy_kernel(const float* __restrict__ src, float* __restrict__ dst,
                                  long long n /*one copy*/, long long total)
{
    long long i = (long long)blockIdx.x * blockDim.x + threadIdx.x;
    if (i < total) dst[i] = src[i % n];
}

__global__ void bias_add_kernel(float* x, const float* __restrict__ b,
                                long long n, int c)
{
    long long i = (long long)blockIdx.x * blockDim.x + threadIdx.x;
    if (i < n) x[i] += b[(int)(i % c)];
}

// ---------------------------------------------------------------------------
// Host orchestration
// ---------------------------------------------------------------------------
struct BNp { const float* g; const float* b; };
struct ResP { const float* w1; BNp bn1; const float* w2; BNp bn2;
              const float* wd; BNp bnd; bool has_wd; };
struct StageP { const float* dw; BNp dbn; ResP r1, r2; };
struct UpP    { const float* dw; BNp dbn; ResP r1, r2; };

static inline unsigned gdim(long long n, int t) { return (unsigned)((n + t - 1) / t); }

extern "C" void kernel_launch(void* const* d_in, const int* in_sizes, int n_in,
                              void* d_out, int out_size, void* d_ws, size_t ws_size,
                              hipStream_t stream)
{
    (void)n_in; (void)out_size; (void)ws_size;
    const int B = 2, N = 8192, IN_DIM = 4, NCLS = 20;
    const int CS[9] = {32, 32, 64, 128, 256, 256, 128, 96, 96};

    int idx = 0;
    // ---- inputs in setup_inputs() dict-insertion order ----
    const float* pt_feats  = (const float*)d_in[idx++];
    const float* pt_coords = (const float*)d_in[idx++];
    // geom: p2v, counts, nbr3[5], nbr2[4], upp[4], upk[4], vmeters[5], seg_idx[4], seg_mask[4]
    const int*   p2v    = (const int*)d_in[idx++];
    const float* counts = (const float*)d_in[idx]; const int M0 = in_sizes[idx]; idx++;
    const int* nbr3[5]; int Ml[5];
    for (int l = 0; l < 5; ++l) { nbr3[l] = (const int*)d_in[idx]; Ml[l] = in_sizes[idx] / 27; idx++; }
    const int* nbr2[4]; for (int s = 0; s < 4; ++s) nbr2[s] = (const int*)d_in[idx++];
    const int* upp[4];  for (int l = 0; l < 4; ++l) upp[l]  = (const int*)d_in[idx++];
    const int* upk[4];  for (int l = 0; l < 4; ++l) upk[l]  = (const int*)d_in[idx++];
    const float* vmet[5]; for (int l = 0; l < 5; ++l) vmet[l] = (const float*)d_in[idx++];
    const int* segidx[4]; int Mmax[4];
    for (int l = 0; l < 4; ++l) { segidx[l] = (const int*)d_in[idx]; Mmax[l] = in_sizes[idx] / B; idx++; }
    const unsigned char* segmask[4];
    for (int l = 0; l < 4; ++l) segmask[l] = (const unsigned char*)d_in[idx++];

    auto getW  = [&]() { return (const float*)d_in[idx++]; };
    auto getBN = [&]() { BNp r; r.g = (const float*)d_in[idx++]; r.b = (const float*)d_in[idx++]; return r; };
    auto getRes = [&](bool has_wd) {
        ResP r; r.w1 = getW(); r.bn1 = getBN(); r.w2 = getW(); r.bn2 = getBN();
        r.has_wd = has_wd; r.wd = nullptr; r.bnd.g = nullptr; r.bnd.b = nullptr;
        if (has_wd) { r.wd = getW(); r.bnd = getBN(); }
        return r;
    };
    // params: stem, stages[4], ups[4], obn[4], head  (insertion order)
    const float* stem_w1 = getW(); BNp stem_bn1 = getBN();
    const float* stem_w2 = getW(); BNp stem_bn2 = getBN();
    StageP st[4];
    for (int s = 0; s < 4; ++s) {
        st[s].dw = getW(); st[s].dbn = getBN();
        st[s].r1 = getRes(CS[s] != CS[s + 1]);
        st[s].r2 = getRes(false);
    }
    UpP up[4];
    for (int u = 0; u < 4; ++u) {
        up[u].dw = getW(); up[u].dbn = getBN();
        up[u].r1 = getRes(true);
        up[u].r2 = getRes(false);
    }
    BNp obn[4]; for (int i = 0; i < 4; ++i) obn[i] = getBN();
    const float* head_w = getW();
    const float* head_b = (const float*)d_in[idx++];

    // ---- workspace bump allocator ----
    char* wsp = (char*)d_ws; size_t wso = 0;
    auto alloc = [&](size_t bytes) -> void* {
        size_t a = (wso + 255) & ~(size_t)255; void* p = wsp + a; wso = a + bytes; return p;
    };
    const int rowsMax = (M0 > B * N) ? M0 : B * N;
    half_t* A16 = (half_t*)alloc((size_t)rowsMax * 384 * sizeof(half_t));
    half_t* Wt  = (half_t*)alloc((size_t)27 * 256 * 384 * sizeof(half_t));
    float* stats = (float*)alloc(2 * 512 * sizeof(float));
    float* v0    = (float*)alloc((size_t)M0 * IN_DIM * sizeof(float));
    float* enc[5];
    for (int l = 0; l < 5; ++l) enc[l] = (float*)alloc((size_t)Ml[l] * CS[l] * sizeof(float));
    float* t1   = (float*)alloc((size_t)rowsMax * 384 * sizeof(float));
    float* t2   = (float*)alloc((size_t)rowsMax * 384 * sizeof(float));
    float* td   = (float*)alloc((size_t)rowsMax * 256 * sizeof(float));
    float* cat  = (float*)alloc((size_t)rowsMax * 384 * sizeof(float));
    float* ybuf = (float*)alloc((size_t)rowsMax * 256 * sizeof(float));
    const int outC[4] = {CS[5], CS[6], CS[7], CS[8]};
    float* outs[4];
    for (int u = 0; u < 4; ++u) outs[u] = (float*)alloc((size_t)Ml[3 - u] * outC[u] * sizeof(float));
    float* gbuf = (float*)alloc((size_t)B * N * 256 * sizeof(float));

    // ---- helpers (all launches on `stream`) ----
    auto gemm = [&](const float* x, int rows_in, int cin,
                    const int* nbr, const int* koff, int K,
                    const float* W, int cout, float* out, int rows_out) {
        int cinP  = (cin + 31) & ~31;
        int coutP = (cout + 31) & ~31;      // 32-wide N-blocking
        conv_a16_kernel<<<gdim((long long)rows_in * cinP, 256), 256, 0, stream>>>(
            x, A16, rows_in, cin, cinP);
        conv_w16_kernel<<<gdim((long long)K * coutP * cinP, 256), 256, 0, stream>>>(
            W, Wt, K, cin, cout, cinP, coutP);
        dim3 grid(gdim(rows_out, 64), coutP / 32);
        dim3 blk(32, 4);
        wmma_gemm_kernel<<<grid, blk, 0, stream>>>(A16, nbr, koff, Wt, out,
                                                   rows_out, cinP, cout, coutP, K);
    };
    auto bn = [&](const float* x, int rows, int c, BNp p, int act,
                  const float* resid, float* out) {
        col_stats_kernel<<<c, 256, 0, stream>>>(x, rows, c, stats);
        bn_apply_kernel<<<gdim((long long)rows * c, 256), 256, 0, stream>>>(
            x, stats, p.g, p.b, resid, out, rows, c, act);
    };
    auto res = [&](const float* x, int rows, int cin, int cout, const int* nbr,
                   const ResP& p, float* out) {
        gemm(x, rows, cin, nbr, nullptr, 27, p.w1, cout, t1, rows);
        bn(t1, rows, cout, p.bn1, 1, nullptr, t1);
        gemm(t1, rows, cout, nbr, nullptr, 27, p.w2, cout, t2, rows);
        const float* dres;
        if (p.has_wd) {
            gemm(x, rows, cin, nullptr, nullptr, 1, p.wd, cout, td, rows);
            bn(td, rows, cout, p.bnd, 0, nullptr, td);
            dres = td;
        } else dres = x;
        bn(t2, rows, cout, p.bn2, 1, dres, out);
    };

    // ---- voxelize (scatter-mean) ----
    zero_kernel<<<gdim((long long)M0 * IN_DIM, 256), 256, 0, stream>>>(v0, (long long)M0 * IN_DIM);
    scatter_mean_kernel<<<gdim(B * N, 256), 256, 0, stream>>>(pt_feats, p2v, v0, B * N, IN_DIM);
    divide_kernel<<<gdim((long long)M0 * IN_DIM, 256), 256, 0, stream>>>(
        v0, counts, (long long)M0 * IN_DIM, IN_DIM);

    // ---- stem ----
    gemm(v0, M0, IN_DIM, nbr3[0], nullptr, 27, stem_w1, CS[0], t1, M0);
    bn(t1, M0, CS[0], stem_bn1, 1, nullptr, t1);
    gemm(t1, M0, CS[0], nbr3[0], nullptr, 27, stem_w2, CS[0], t2, M0);
    bn(t2, M0, CS[0], stem_bn2, 1, nullptr, enc[0]);

    // ---- encoder stages ----
    const float* x = enc[0];
    for (int s = 0; s < 4; ++s) {
        int cprev = CS[s], cnew = CS[s + 1];
        int rin = Ml[s], rout = Ml[s + 1];
        gemm(x, rin, cprev, nbr2[s], nullptr, 8, st[s].dw, cprev, t1, rout);
        bn(t1, rout, cprev, st[s].dbn, 2, nullptr, ybuf);        // leaky 0.01
        res(ybuf, rout, cprev, cnew, nbr3[s + 1], st[s].r1, cat);
        res(cat, rout, cnew, cnew, nbr3[s + 1], st[s].r2, enc[s + 1]);
        x = enc[s + 1];
    }

    // ---- decoder ----
    const float* y = enc[4];
    int cy = CS[4];
    for (int u = 0; u < 4; ++u) {
        int l = 3 - u;
        int cyo = CS[5 + u], csk = CS[l];
        int rows = Ml[l], rin = Ml[l + 1];
        gemm(y, rin, cy, upp[l], upk[l], 8, up[u].dw, cyo, t1, rows);
        bn(t1, rows, cyo, up[u].dbn, 2, nullptr, t1);
        copy_cols_kernel<<<gdim((long long)rows * cyo, 256), 256, 0, stream>>>(
            t1, cat, rows, cyo, cyo + csk, 0);
        copy_cols_kernel<<<gdim((long long)rows * csk, 256), 256, 0, stream>>>(
            enc[l], cat, rows, csk, cyo + csk, cyo);
        res(cat, rows, cyo + csk, cyo, nbr3[l], up[u].r1, ybuf);
        res(ybuf, rows, cyo, cyo, nbr3[l], up[u].r2, outs[u]);
        y = outs[u]; cy = cyo;
    }

    // ---- per-point interpolation + per-batch BN -> feats in d_out ----
    float* outf = (float*)d_out;
    long long outoff = 0;
    for (int i = 0; i < 4; ++i) {
        int l = 3 - i, C = outC[i];
        knn_interp_kernel<<<gdim(B * N, 256), 256, 0, stream>>>(
            pt_coords, vmet[l], outs[i], segidx[l], segmask[l], N, Mmax[l], C, B, gbuf);
        for (int b = 0; b < B; ++b) {
            const float* slice = gbuf + (long long)b * N * C;
            col_stats_kernel<<<C, 256, 0, stream>>>(slice, N, C, stats);
            bn_apply_kernel<<<gdim((long long)N * C, 256), 256, 0, stream>>>(
                slice, stats, obn[i].g, obn[i].b, nullptr,
                outf + outoff + (long long)b * N * C, N, C, 0);
        }
        outoff += (long long)B * N * C;
    }

    // ---- coors (4 copies of pt_coords) + pad (zeros) ----
    long long ncoor = (long long)B * N * 3;
    coors_copy_kernel<<<gdim(4 * ncoor, 256), 256, 0, stream>>>(
        pt_coords, outf + outoff, ncoor, 4 * ncoor);
    outoff += 4 * ncoor;
    zero_kernel<<<gdim((long long)4 * B * N, 256), 256, 0, stream>>>(
        outf + outoff, (long long)4 * B * N);
    outoff += (long long)4 * B * N;

    // ---- head: logits = feats[3] @ W + b ----
    const float* f3 = outf + (long long)B * N * (CS[5] + CS[6] + CS[7]);
    gemm(f3, B * N, CS[8], nullptr, nullptr, 1, head_w, NCLS, outf + outoff, B * N);
    bias_add_kernel<<<gdim((long long)B * N * NCLS, 256), 256, 0, stream>>>(
        outf + outoff, head_b, (long long)B * N * NCLS, NCLS);
}
